// ParticleGNNBigger_25812753449154
// MI455X (gfx1250) — matrix operand order repaired and checked by reading, not compiled, not run on hardware
//
#include <hip/hip_runtime.h>
#include <hip/hip_bf16.h>
#include <math.h>

typedef __attribute__((ext_vector_type(16))) __bf16 v16bf;
typedef __attribute__((ext_vector_type(8)))  __bf16 v8bf;
typedef __attribute__((ext_vector_type(8)))  float  v8f;

#define NN   50000
#define EE   400000
#define ETOT (EE + NN)

// ---------------------------------------------------------------------------
// small helpers
// ---------------------------------------------------------------------------
static __device__ __forceinline__ void atomicMaxF(float* a, float v) {
  // works for all finite floats (pos path: int max, neg path: uint min)
  if (v >= 0.0f) atomicMax((int*)a, __float_as_int(v));
  else           atomicMin((unsigned int*)a, __float_as_uint(v));
}

__global__ void fill_f32(float* p, float v, long n) {
  long i = (long)blockIdx.x * blockDim.x + threadIdx.x;
  long st = (long)gridDim.x * blockDim.x;
  for (; i < n; i += st) p[i] = v;
}

__global__ void cvt_bf16(__bf16* dst, const float* src, long n) {
  long i = (long)blockIdx.x * blockDim.x + threadIdx.x;
  long st = (long)gridDim.x * blockDim.x;
  for (; i < n; i += st) dst[i] = (__bf16)src[i];
}

// W [K, Nout] f32 (row-major)  ->  Wt [Nout, K] bf16 (row-major)
__global__ void transpose_cvt(__bf16* wt, const float* w, int K, int Nout) {
  long n = (long)K * Nout;
  long st = (long)gridDim.x * blockDim.x;
  for (long i = (long)blockIdx.x * blockDim.x + threadIdx.x; i < n; i += st) {
    int no = (int)(i / K);
    int k  = (int)(i % K);
    wt[i] = (__bf16)w[(size_t)k * Nout + no];
  }
}

// ---------------------------------------------------------------------------
// WMMA GEMM: C[M,Nout] = A[M,K](bf16) * Wt[Nout,K]^T(bf16), f32 accumulate.
// K is compile-time: the K-loop fully unrolls with no guards / EXEC masking.
// Each wave computes one 16x32 output tile (2 adjacent N-tiles sharing the A
// fragment -> 2 independent v_wmma chains per iteration). blockDim = (32,2).
// bf16 epilogue stages the tile in LDS and writes coalesced b128 stores.
// Fragment layouts per CDNA5 ISA 7.12.2 (16-bit A 16x32, B 32x16, f32 C/D).
// ---------------------------------------------------------------------------
template <int K, bool OUT_BF16>
__global__ __launch_bounds__(64) void wmma_gemm_k(const __bf16* __restrict__ A,
                                                  const __bf16* __restrict__ Bt,
                                                  void* __restrict__ Cout,
                                                  int Nout) {
  const int lane = threadIdx.x;        // 0..31
  const int half = lane >> 4;          // 0/1
  const int l15  = lane & 15;
  const int tm   = blockIdx.x;
  const int tn0  = (blockIdx.y * 2 + threadIdx.y) * 2;  // first of 2 N-tiles

  const __bf16* arow  = A  + (size_t)(tm * 16 + l15) * K;
  const __bf16* brow0 = Bt + (size_t)(tn0 * 16 + l15) * K;
  const __bf16* brow1 = brow0 + (size_t)16 * K;

  v8f acc0 = {};
  v8f acc1 = {};
#pragma unroll
  for (int k0 = 0; k0 < K; k0 += 32) {
    v16bf av = {}, bv0 = {}, bv1 = {};
    // A fragment: lanes 0-15 K={k0..k0+7, k0+16..k0+23}; lanes 16-31 +8
    const int ak = k0 + half * 8;
    {
      v8bf lo = *(const v8bf*)(arow + ak);
#pragma unroll
      for (int i = 0; i < 8; ++i) av[i] = lo[i];
      if constexpr (K % 32 == 0) {
        v8bf hi = *(const v8bf*)(arow + ak + 16);
#pragma unroll
        for (int i = 0; i < 8; ++i) av[8 + i] = hi[i];
      }
    }
    // B fragment: lanes 0-15 K={k0..k0+7, k0+16..k0+23}; lanes 16-31 +8
    const int bk = k0 + half * 8;
    if constexpr (K % 32 == 0) {
      v8bf p0 = *(const v8bf*)(brow0 + bk);
      v8bf p1 = *(const v8bf*)(brow0 + bk + 16);
      v8bf q0 = *(const v8bf*)(brow1 + bk);
      v8bf q1 = *(const v8bf*)(brow1 + bk + 16);
#pragma unroll
      for (int i = 0; i < 8; ++i) {
        bv0[i] = p0[i]; bv0[8 + i] = p1[i];
        bv1[i] = q0[i]; bv1[8 + i] = q1[i];
      }
    } else {
      // K == 16: only elements 0-7 of each half-wave exist; 8-15 stay zero.
      {
        v8bf p0 = *(const v8bf*)(brow0 + bk);
        v8bf q0 = *(const v8bf*)(brow1 + bk);
#pragma unroll
        for (int i = 0; i < 8; ++i) {
          bv0[i] = p0[i];
          bv1[i] = q0[i];
        }
      }
    }
    acc0 = __builtin_amdgcn_wmma_f32_16x16x32_bf16(
        false, av, false, bv0, (short)0, acc0, false, false);
    acc1 = __builtin_amdgcn_wmma_f32_16x16x32_bf16(
        false, av, false, bv1, (short)0, acc1, false, false);
  }

  // D layout: lane -> col = lane&15 ; VGPR r -> row (lane>=16 ? 8 : 0)+r
  if constexpr (OUT_BF16) {
    // Stage 16x32 bf16 tile in LDS, then coalesced b128 stores (2 per lane).
    __shared__ __bf16 lds[2][16 * 32];
    __bf16* tile = lds[threadIdx.y];
#pragma unroll
    for (int r = 0; r < 8; ++r) {
      tile[(half * 8 + r) * 32 + l15]      = (__bf16)acc0[r];
      tile[(half * 8 + r) * 32 + l15 + 16] = (__bf16)acc1[r];
    }
    __syncthreads();
    const int rrow = lane >> 1;           // 0..15
    const int rcol = (lane & 1) * 16;     // 0 or 16
    const __bf16* src = tile + rrow * 32 + rcol;
    v8bf c0 = *(const v8bf*)(src);
    v8bf c1 = *(const v8bf*)(src + 8);
    __bf16* dst = (__bf16*)Cout + (size_t)(tm * 16 + rrow) * Nout + tn0 * 16 + rcol;
    *(v8bf*)(dst)     = c0;
    *(v8bf*)(dst + 8) = c1;
  } else {
    const int col0 = tn0 * 16 + l15;
    const int m0   = tm * 16 + half * 8;
#pragma unroll
    for (int r = 0; r < 8; ++r) {
      size_t i0 = (size_t)(m0 + r) * Nout + col0;
      ((float*)Cout)[i0]      = acc0[r];
      ((float*)Cout)[i0 + 16] = acc1[r];
    }
  }
}

// ---------------------------------------------------------------------------
// edge-attr mean (reduction) and per-head edge scalar s[h]
// ---------------------------------------------------------------------------
__global__ void reduce_sum(const float* x, long n, float* out) {
  __shared__ float sh[256];
  float s = 0.0f;
  long st = (long)gridDim.x * blockDim.x;
  for (long i = (long)blockIdx.x * blockDim.x + threadIdx.x; i < n; i += st) s += x[i];
  sh[threadIdx.x] = s;
  __syncthreads();
  for (int off = 128; off > 0; off >>= 1) {
    if ((int)threadIdx.x < off) sh[threadIdx.x] += sh[threadIdx.x + off];
    __syncthreads();
  }
  if (threadIdx.x == 0) atomicAdd(out, sh[0]);
}

__global__ void finalize_mean(float* stats, float inv) {
  if (threadIdx.x == 0 && blockIdx.x == 0) stats[1] = stats[0] * inv;
}

// s[h] = sum_c W_edge[h*C+c] * att_edge[h*C+c]
__global__ void edge_scalar(const float* W_edge, const float* att_edge,
                            int H, int C, float* sbuf) {
  int h = threadIdx.x;
  if (h < H) {
    float s = 0.0f;
    for (int c = 0; c < C; ++c) s += W_edge[h * C + c] * att_edge[h * C + c];
    sbuf[h] = s;
  }
}

// ---------------------------------------------------------------------------
// attention node dots: a_src[n,h], a_dst[n,h] from bf16 h_lin
// ---------------------------------------------------------------------------
__global__ void attn_node(const __bf16* __restrict__ hlin,
                          const float* __restrict__ att_src,
                          const float* __restrict__ att_dst,
                          int H, int C, float* asrc, float* adst) {
  long i = (long)blockIdx.x * blockDim.x + threadIdx.x;
  if (i >= (long)NN * H) return;
  int n = (int)(i / H), h = (int)(i % H);
  const __bf16* hp = hlin + (size_t)n * H * C + (size_t)h * C;
  float s1 = 0.0f, s2 = 0.0f;
  for (int c = 0; c < C; ++c) {
    float v = (float)hp[c];
    s1 += v * att_src[h * C + c];
    s2 += v * att_dst[h * C + c];
  }
  asrc[i] = s1;
  adst[i] = s2;
}

static __device__ __forceinline__ void edge_sd(const int* ei, long e, int& s, int& d) {
  if (e < EE) { s = ei[e]; d = ei[EE + e]; }
  else        { s = d = (int)(e - EE); }
}

// alpha = leaky_relu(a_src[s]+a_dst[d]+ea*s_h); store; segment max via atomics
__global__ void edge_alpha(const int* __restrict__ ei,
                           const float* __restrict__ edge_attr,
                           const float* __restrict__ stats,  // [1] = ea mean
                           const float* __restrict__ sbuf,
                           const float* __restrict__ asrc,
                           const float* __restrict__ adst,
                           int H, float* alphabuf, float* amax) {
  long i = (long)blockIdx.x * blockDim.x + threadIdx.x;
  if (i >= (long)ETOT * H) return;
  long e = i / H; int h = (int)(i % H);
  int s, d; edge_sd(ei, e, s, d);
  float ea = (e < EE) ? edge_attr[e] : stats[1];
  float a = asrc[(size_t)s * H + h] + adst[(size_t)d * H + h] + ea * sbuf[h];
  a = a > 0.0f ? a : 0.2f * a;
  alphabuf[i] = a;
  atomicMaxF(&amax[(size_t)d * H + h], a);
}

__global__ void edge_exp(const int* __restrict__ ei,
                         const float* __restrict__ amax,
                         int H, float* alphabuf, float* denom) {
  long i = (long)blockIdx.x * blockDim.x + threadIdx.x;
  if (i >= (long)ETOT * H) return;
  long e = i / H; int h = (int)(i % H);
  int s, d; edge_sd(ei, e, s, d);
  float ex = __expf(alphabuf[i] - amax[(size_t)d * H + h]);
  alphabuf[i] = ex;
  atomicAdd(&denom[(size_t)d * H + h], ex);
}

// out[d,h,:] += w * h_lin[s,h,:]
__global__ void edge_scatter(const int* __restrict__ ei,
                             const __bf16* __restrict__ hlin,
                             const float* __restrict__ exbuf,
                             const float* __restrict__ denom,
                             int H, int C, float* agg) {
  long i = (long)blockIdx.x * blockDim.x + threadIdx.x;
  if (i >= (long)ETOT * H) return;
  long e = i / H; int h = (int)(i % H);
  int s, d; edge_sd(ei, e, s, d);
  float w = exbuf[i] / (denom[(size_t)d * H + h] + 1e-16f);
  const __bf16* hp = hlin + (size_t)s * H * C + (size_t)h * C;
  float* op = agg + (size_t)d * H * C + (size_t)h * C;
  for (int c = 0; c < C; ++c) atomicAdd(&op[c], w * (float)hp[c]);
}

// ---------------------------------------------------------------------------
// training-mode BatchNorm: one block per channel (no atomics), then fused
// normalize+ReLU elementwise.
// ---------------------------------------------------------------------------
__global__ void bn_stats(const float* __restrict__ x, int n, int D,
                         float* meanv, float* varv) {
  int c = blockIdx.x;
  float s = 0.0f, s2 = 0.0f;
  for (int r = threadIdx.x; r < n; r += blockDim.x) {
    float v = x[(size_t)r * D + c];
    s += v; s2 += v * v;
  }
  __shared__ float sh[256], sh2[256];
  sh[threadIdx.x] = s; sh2[threadIdx.x] = s2;
  __syncthreads();
  for (int off = 128; off > 0; off >>= 1) {
    if ((int)threadIdx.x < off) {
      sh[threadIdx.x]  += sh[threadIdx.x + off];
      sh2[threadIdx.x] += sh2[threadIdx.x + off];
    }
    __syncthreads();
  }
  if (threadIdx.x == 0) {
    float m = sh[0] / n;
    meanv[c] = m;
    varv[c]  = sh2[0] / n - m * m;
  }
}

__global__ void bn_relu(float* x, const float* g, const float* b,
                        const float* meanv, const float* varv,
                        long n, int D) {
  long i = (long)blockIdx.x * blockDim.x + threadIdx.x;
  long st = (long)gridDim.x * blockDim.x;
  for (; i < n; i += st) {
    int c = (int)(i % D);
    float y = g[c] * (x[i] - meanv[c]) * __frsqrt_rn(varv[c] + 1e-5f) + b[c];
    x[i] = y > 0.0f ? y : 0.0f;
  }
}

// final 64->3 linear + log_softmax
__global__ void mlp_final(const float* __restrict__ h,
                          const float* __restrict__ W3,
                          const float* __restrict__ b3,
                          float* out) {
  int n = blockIdx.x * blockDim.x + threadIdx.x;
  if (n >= NN) return;
  float z0 = b3[0], z1 = b3[1], z2 = b3[2];
  const float* hp = h + (size_t)n * 64;
  for (int k = 0; k < 64; ++k) {
    float v = hp[k];
    z0 += v * W3[k * 3 + 0];
    z1 += v * W3[k * 3 + 1];
    z2 += v * W3[k * 3 + 2];
  }
  float m = fmaxf(z0, fmaxf(z1, z2));
  float lse = m + __logf(__expf(z0 - m) + __expf(z1 - m) + __expf(z2 - m));
  out[n * 3 + 0] = z0 - lse;
  out[n * 3 + 1] = z1 - lse;
  out[n * 3 + 2] = z2 - lse;
}

// ---------------------------------------------------------------------------
// driver
// ---------------------------------------------------------------------------
extern "C" void kernel_launch(void* const* d_in, const int* in_sizes, int n_in,
                              void* d_out, int out_size, void* d_ws, size_t ws_size,
                              hipStream_t stream) {
  (void)in_sizes; (void)n_in; (void)out_size; (void)ws_size;

  const float* x         = (const float*)d_in[0];
  const int*   ei        = (const int*)d_in[1];
  const float* edge_attr = (const float*)d_in[2];
  auto convP = [&](int i, int j) { return (const float*)d_in[3 + 6 * i + j]; };
  const int bnsBase = 3 + 36;       // 39
  const int mlpBase = bnsBase + 12; // 51
  auto bnG = [&](int i) { return (const float*)d_in[bnsBase + 2 * i + 0]; };
  auto bnB = [&](int i) { return (const float*)d_in[bnsBase + 2 * i + 1]; };
  auto mlpP = [&](int j) { return (const float*)d_in[mlpBase + j]; };

  // workspace carve-up
  char* ws = (char*)d_ws;
  size_t off = 0;
  auto carve = [&](size_t bytes) {
    size_t r = off;
    off += (bytes + 255) & ~(size_t)255;
    return (void*)(ws + r);
  };
  float*  buf_h   = (float*)carve((size_t)NN * 512 * 4);   // h_cur / agg target
  __bf16* buf_lin = (__bf16*)carve((size_t)NN * 512 * 2);  // h_lin (bf16) / MLP f32 temp
  __bf16* buf_hbf = (__bf16*)carve((size_t)NN * 512 * 2);  // bf16 GEMM A
  __bf16* buf_wt  = (__bf16*)carve((size_t)512 * 512 * 2); // bf16 W^T
  float*  asrc    = (float*)carve((size_t)NN * 4 * 4);
  float*  adst    = (float*)carve((size_t)NN * 4 * 4);
  float*  amax    = (float*)carve((size_t)NN * 4 * 4);
  float*  denom   = (float*)carve((size_t)NN * 4 * 4);
  float*  alphab  = (float*)carve((size_t)ETOT * 4 * 4);
  float*  meanv   = (float*)carve(512 * 4);
  float*  varv    = (float*)carve(512 * 4);
  float*  stats   = (float*)carve(256);
  float*  sbuf    = (float*)carve(256);

  const int T = 256;
  auto blocks = [&](long n) { return (int)((n + T - 1) / T); };

  // GEMM dispatch on compile-time K (one wave = 16x32 tile; block = 2 waves
  // = 16x64; grid.y = Nout/64, exact for Nout in {64,128,256,512}).
  auto gemm = [&](const __bf16* Abf, void* Cout, int K, int Nout, bool outbf) {
    dim3 g(NN / 16, Nout / 64), b(32, 2);
    switch (K) {
      case 16:
        if (outbf) wmma_gemm_k<16,  true ><<<g, b, 0, stream>>>(Abf, buf_wt, Cout, Nout);
        else       wmma_gemm_k<16,  false><<<g, b, 0, stream>>>(Abf, buf_wt, Cout, Nout);
        break;
      case 64:
        if (outbf) wmma_gemm_k<64,  true ><<<g, b, 0, stream>>>(Abf, buf_wt, Cout, Nout);
        else       wmma_gemm_k<64,  false><<<g, b, 0, stream>>>(Abf, buf_wt, Cout, Nout);
        break;
      case 128:
        if (outbf) wmma_gemm_k<128, true ><<<g, b, 0, stream>>>(Abf, buf_wt, Cout, Nout);
        else       wmma_gemm_k<128, false><<<g, b, 0, stream>>>(Abf, buf_wt, Cout, Nout);
        break;
      case 256:
        if (outbf) wmma_gemm_k<256, true ><<<g, b, 0, stream>>>(Abf, buf_wt, Cout, Nout);
        else       wmma_gemm_k<256, false><<<g, b, 0, stream>>>(Abf, buf_wt, Cout, Nout);
        break;
      case 512:
        if (outbf) wmma_gemm_k<512, true ><<<g, b, 0, stream>>>(Abf, buf_wt, Cout, Nout);
        else       wmma_gemm_k<512, false><<<g, b, 0, stream>>>(Abf, buf_wt, Cout, Nout);
        break;
    }
  };

  // edge_attr mean (self-loop fill value)
  fill_f32<<<1, T, 0, stream>>>(stats, 0.0f, 2);
  reduce_sum<<<512, T, 0, stream>>>(edge_attr, EE, stats);
  finalize_mean<<<1, 32, 0, stream>>>(stats, 1.0f / (float)EE);

  static const int CIN[6] = {16, 256, 256, 512, 512, 256};
  static const int HH [6] = {4, 4, 4, 4, 4, 1};
  static const int CC [6] = {64, 64, 128, 128, 64, 64};

  for (int i = 0; i < 6; ++i) {
    const int cin = CIN[i], H = HH[i], C = CC[i], HC = H * C;

    // linear: h_lin = h_cur @ W   (bf16 in / bf16 out, f32 accumulate via WMMA)
    transpose_cvt<<<blocks((long)cin * HC), T, 0, stream>>>(buf_wt, convP(i, 0), cin, HC);
    cvt_bf16<<<blocks((long)NN * cin), T, 0, stream>>>(
        buf_hbf, (i == 0) ? x : buf_h, (long)NN * cin);
    gemm(buf_hbf, buf_lin, cin, HC, true);

    // attention scalars
    edge_scalar<<<1, 32, 0, stream>>>(convP(i, 3), convP(i, 4), H, C, sbuf);
    attn_node<<<blocks((long)NN * H), T, 0, stream>>>(
        buf_lin, convP(i, 1), convP(i, 2), H, C, asrc, adst);

    // segment softmax + weighted scatter-add
    fill_f32<<<blocks((long)NN * H), T, 0, stream>>>(amax, -1e30f, (long)NN * H);
    fill_f32<<<blocks((long)NN * H), T, 0, stream>>>(denom, 0.0f, (long)NN * H);
    fill_f32<<<2048, T, 0, stream>>>(buf_h, 0.0f, (long)NN * HC);
    edge_alpha<<<blocks((long)ETOT * H), T, 0, stream>>>(
        ei, edge_attr, stats, sbuf, asrc, adst, H, alphab, amax);
    edge_exp<<<blocks((long)ETOT * H), T, 0, stream>>>(ei, amax, H, alphab, denom);
    edge_scatter<<<blocks((long)ETOT * H), T, 0, stream>>>(
        ei, buf_lin, alphab, denom, H, C, buf_h);

    // BatchNorm (bias before BN is a no-op in training mode) + ReLU
    bn_stats<<<HC, T, 0, stream>>>(buf_h, NN, HC, meanv, varv);
    bn_relu<<<blocks((long)NN * HC), T, 0, stream>>>(
        buf_h, bnG(i), bnB(i), meanv, varv, (long)NN * HC, HC);
  }

  // ---- MLP head ----
  float* h1 = (float*)buf_lin; // N x 128 f32 fits in the 51MB bf16 buffer

  // fc1: 64 -> 128, BN(g1,b1n)+ReLU   (b1 cancels in BN)
  transpose_cvt<<<blocks(64L * 128), T, 0, stream>>>(buf_wt, mlpP(0), 64, 128);
  cvt_bf16<<<blocks((long)NN * 64), T, 0, stream>>>(buf_hbf, buf_h, (long)NN * 64);
  gemm(buf_hbf, h1, 64, 128, false);
  bn_stats<<<128, T, 0, stream>>>(h1, NN, 128, meanv, varv);
  bn_relu<<<blocks((long)NN * 128), T, 0, stream>>>(
      h1, mlpP(2), mlpP(3), meanv, varv, (long)NN * 128, 128);

  // fc2: 128 -> 64, BN(g2,b2n)+ReLU
  transpose_cvt<<<blocks(128L * 64), T, 0, stream>>>(buf_wt, mlpP(4), 128, 64);
  cvt_bf16<<<blocks((long)NN * 128), T, 0, stream>>>(buf_hbf, h1, (long)NN * 128);
  gemm(buf_hbf, buf_h, 128, 64, false);
  bn_stats<<<64, T, 0, stream>>>(buf_h, NN, 64, meanv, varv);
  bn_relu<<<blocks((long)NN * 64), T, 0, stream>>>(
      buf_h, mlpP(6), mlpP(7), meanv, varv, (long)NN * 64, 64);

  // fc3: 64 -> 3 + log_softmax
  mlp_final<<<blocks(NN), T, 0, stream>>>(buf_h, mlpP(8), mlpP(9), (float*)d_out);
}
